// NS_aggr_3221225472214
// MI455X (gfx1250) — matrix operand order repaired and checked by reading, not compiled, not run on hardware
//
#include <hip/hip_runtime.h>

typedef __attribute__((ext_vector_type(16))) __bf16 v16bf;
typedef __attribute__((ext_vector_type(8)))  __bf16 v8bf;
typedef __attribute__((ext_vector_type(8)))  float  v8f;

#define NODE_NUM 65536
#define HID 256
#define NUM_CLASS 64
#define N_TAB0 40000
#define N_TAB1 30000
#define ROWS_TOTAL (N_TAB0 + N_TAB1 + NODE_NUM)   /* 135536 */
#define ROW_TILES (ROWS_TOTAL / 16)               /* 8471, exact */
#define WAVES_PER_BLOCK 4

__device__ __forceinline__ __bf16 f2bf(float f) {
  unsigned u = __builtin_bit_cast(unsigned, f);
  u += 0x7FFFu + ((u >> 16) & 1u);               // round to nearest even
  unsigned short h = (unsigned short)(u >> 16);
  return __builtin_bit_cast(__bf16, h);
}

// ---------------------------------------------------------------------------
// Kernel 0: convert W1 (256x256 f32) and W2 (256x64 f32) into bf16 stored in
// the WMMA B-fragment layout, so the GEMM kernel loads each lane's 16 values
// with one contiguous 32-byte load.
// B layout (16-bit, 32x16 tile): lane l -> n = l&15, k = (l<16?0:8)+(e<8?e:e+8), e=0..15
// flat index = ((nt*8 + kt)*32 + l)*16 + e
// ---------------------------------------------------------------------------
__global__ __launch_bounds__(256) void convert_weights_kernel(
    const float* __restrict__ W1, const float* __restrict__ W2,
    __bf16* __restrict__ W1b, __bf16* __restrict__ W2b)
{
  int i = blockIdx.x * 256 + threadIdx.x;
  if (i < 256 * 256) {                       // W1: 16 n-tiles x 8 k-tiles
    int e  = i & 15;
    int l  = (i >> 4) & 31;
    int kt = (i >> 9) & 7;
    int nt = i >> 12;
    int k  = kt * 32 + ((l < 16) ? 0 : 8) + ((e < 8) ? e : e + 8);
    int n  = nt * 16 + (l & 15);
    W1b[i] = f2bf(W1[k * HID + n]);
  }
  if (i < 256 * 64) {                        // W2: 4 n-tiles x 8 k-tiles
    int e  = i & 15;
    int l  = (i >> 4) & 31;
    int kt = (i >> 9) & 7;
    int nt = i >> 12;
    int k  = kt * 32 + ((l < 16) ? 0 : 8) + ((e < 8) ? e : e + 8);
    int n  = nt * 16 + (l & 15);
    W2b[i] = f2bf(W2[k * NUM_CLASS + n]);
  }
}

// ---------------------------------------------------------------------------
// Kernel 1: fused MLP(256->256 relu ->64) + softmax over the 135536 unique
// rows [tab0; tab1; h0]. One wave per 16-row tile, bf16 WMMA, f32 accum.
// Layer 1 runs two independent accumulator chains per loop iteration so the
// WMMA pipeline never serializes on a single C->D chain.
// ---------------------------------------------------------------------------
__global__ __launch_bounds__(128) void mlp_softmax_kernel(
    const float* __restrict__ h0, const float* __restrict__ tab0,
    const float* __restrict__ tab1,
    const __bf16* __restrict__ W1b, const __bf16* __restrict__ W2b,
    const float* __restrict__ b1, const float* __restrict__ b2,
    float* __restrict__ probs)
{
  __shared__ __bf16 Ylds[WAVES_PER_BLOCK][16 * HID];   // 32 KB, wave-private slabs

  const int wave = threadIdx.x >> 5;
  const int lane = threadIdx.x & 31;
  const int tile = blockIdx.x * WAVES_PER_BLOCK + wave;
  if (tile >= ROW_TILES) return;                       // no block-wide sync used

  const int row_base = tile * 16;
  const float* src; int srow;
  if (row_base < N_TAB0)               { src = tab0; srow = row_base; }
  else if (row_base < N_TAB0 + N_TAB1) { src = tab1; srow = row_base - N_TAB0; }
  else                                 { src = h0;   srow = row_base - (N_TAB0 + N_TAB1); }

  const int m  = lane & 15;                 // row within tile this lane feeds
  const int kb = (lane < 16) ? 0 : 8;       // A-layout K sub-base within 32-group
  const float* xrow = src + (size_t)(srow + m) * HID;

  // ---- build A fragments (16x32 bf16 each) for all 8 K-steps (K = 256)
  // lane l holds K = kb..kb+7 and kb+16..kb+23 within each 32-wide K group
  v16bf A[8];
  #pragma unroll
  for (int kt = 0; kt < 8; ++kt) {
    const float* p0 = xrow + kt * 32 + kb;
    #pragma unroll
    for (int e = 0; e < 8; ++e) {
      A[kt][e]     = f2bf(p0[e]);
      A[kt][e + 8] = f2bf(p0[e + 16]);
    }
  }

  __bf16* y = &Ylds[wave][0];

  // ---- layer 1: Y[16,256] = relu(X @ W1 + b1), stored to LDS as bf16
  for (int nt = 0; nt < 16; nt += 2) {
    v8f acc0 = {};
    v8f acc1 = {};
    #pragma unroll
    for (int kt = 0; kt < 8; ++kt) {
      v16bf B0 = *(const v16bf*)(W1b + ((((nt + 0) * 8 + kt) * 32 + lane) << 4));
      v16bf B1 = *(const v16bf*)(W1b + ((((nt + 1) * 8 + kt) * 32 + lane) << 4));
      acc0 = __builtin_amdgcn_wmma_f32_16x16x32_bf16(
          false, A[kt], false, B0, (short)0, acc0, false, false);
      acc1 = __builtin_amdgcn_wmma_f32_16x16x32_bf16(
          false, A[kt], false, B1, (short)0, acc1, false, false);
    }
    const float bias0 = b1[(nt + 0) * 16 + (lane & 15)];
    const float bias1 = b1[(nt + 1) * 16 + (lane & 15)];
    const int   nn0   = (nt + 0) * 16 + (lane & 15);
    const int   nn1   = (nt + 1) * 16 + (lane & 15);
    #pragma unroll
    for (int v = 0; v < 8; ++v) {
      const int mm = v + ((lane < 16) ? 0 : 8);   // D layout: row = vgpr + half*8
      y[mm * HID + nn0] = f2bf(fmaxf(acc0[v] + bias0, 0.0f));
      y[mm * HID + nn1] = f2bf(fmaxf(acc1[v] + bias1, 0.0f));
    }
  }
  asm volatile("s_wait_dscnt 0" ::: "memory");    // wave-local LDS RAW fence

  // ---- layer 2: logits[16,64] = Y @ W2 + b2 (4 independent chains)
  v8f acc2[4];
  #pragma unroll
  for (int nt = 0; nt < 4; ++nt) acc2[nt] = (v8f){};

  for (int kt = 0; kt < 8; ++kt) {
    const __bf16* pr = y + m * HID + kt * 32 + kb;
    v8bf lo = *(const v8bf*)pr;
    v8bf hi = *(const v8bf*)(pr + 16);
    v16bf A2;
    #pragma unroll
    for (int e = 0; e < 8; ++e) { A2[e] = lo[e]; A2[e + 8] = hi[e]; }
    #pragma unroll
    for (int nt = 0; nt < 4; ++nt) {
      v16bf B = *(const v16bf*)(W2b + (((nt * 8 + kt) * 32 + lane) << 4));
      acc2[nt] = __builtin_amdgcn_wmma_f32_16x16x32_bf16(
          false, A2, false, B, (short)0, acc2[nt], false, false);
    }
  }
  #pragma unroll
  for (int nt = 0; nt < 4; ++nt) {
    const float bias = b2[nt * 16 + (lane & 15)];
    #pragma unroll
    for (int v = 0; v < 8; ++v) acc2[nt][v] += bias;
  }

  // ---- softmax over 64 classes per row, rows live in 16-lane halves
  #pragma unroll
  for (int v = 0; v < 8; ++v) {
    float mx = acc2[0][v];
    #pragma unroll
    for (int nt = 1; nt < 4; ++nt) mx = fmaxf(mx, acc2[nt][v]);
    #pragma unroll
    for (int off = 8; off >= 1; off >>= 1)
      mx = fmaxf(mx, __shfl_xor(mx, off, 32));    // stays inside 16-lane half

    float ex[4]; float s = 0.0f;
    #pragma unroll
    for (int nt = 0; nt < 4; ++nt) { ex[nt] = __expf(acc2[nt][v] - mx); s += ex[nt]; }
    #pragma unroll
    for (int off = 8; off >= 1; off >>= 1)
      s += __shfl_xor(s, off, 32);
    const float inv = 1.0f / s;

    const int mm = v + ((lane < 16) ? 0 : 8);
    float* orow = probs + (size_t)(row_base + mm) * NUM_CLASS;
    #pragma unroll
    for (int nt = 0; nt < 4; ++nt)
      orow[nt * 16 + (lane & 15)] = ex[nt] * inv;
  }
}

// ---------------------------------------------------------------------------
// Kernel 2: per-node gather of probability rows (L2-resident) + blend.
// 64 threads per node, thread t -> class c = t & 63.
// ---------------------------------------------------------------------------
__global__ __launch_bounds__(256) void combine_kernel(
    const float* __restrict__ probs,
    const int* __restrict__ idx0, const int* __restrict__ idx1,
    const float* __restrict__ alpha, float* __restrict__ out)
{
  const int t    = blockIdx.x * 256 + threadIdx.x;
  const int node = t >> 6;
  const int c    = t & 63;
  if (node >= NODE_NUM) return;

  const float* p0 = probs;                                         // [40000,64]
  const float* p1 = probs + (size_t)N_TAB0 * NUM_CLASS;            // [30000,64]
  const float* ph = probs + (size_t)(N_TAB0 + N_TAB1) * NUM_CLASS; // [65536,64]

  float s0 = 0.0f;
  #pragma unroll
  for (int s = 0; s < 8; ++s)
    s0 += p0[(size_t)idx0[node * 8 + s] * NUM_CLASS + c];
  float s1 = 0.0f;
  #pragma unroll
  for (int s = 0; s < 4; ++s)
    s1 += p1[(size_t)idx1[node * 4 + s] * NUM_CLASS + c];

  const float nei = 0.5f * (s0 * 0.125f + s1 * 0.25f);
  const float a   = 1.0f / (1.0f + __expf(-alpha[node]));
  out[(size_t)node * NUM_CLASS + c] =
      a * ph[(size_t)node * NUM_CLASS + c] + (1.0f - a) * nei;
}

// ---------------------------------------------------------------------------
extern "C" void kernel_launch(void* const* d_in, const int* in_sizes, int n_in,
                              void* d_out, int out_size, void* d_ws, size_t ws_size,
                              hipStream_t stream) {
  const float* h0    = (const float*)d_in[0];
  const float* tab0  = (const float*)d_in[1];
  const float* tab1  = (const float*)d_in[2];
  const int*   idx0  = (const int*)  d_in[3];
  const int*   idx1  = (const int*)  d_in[4];
  const float* W1    = (const float*)d_in[5];
  const float* b1    = (const float*)d_in[6];
  const float* W2    = (const float*)d_in[7];
  const float* b2    = (const float*)d_in[8];
  const float* alpha = (const float*)d_in[9];
  float* out = (float*)d_out;

  // workspace layout: W1b (128KB) | W2b (32KB) | probs (135536*64 f32)
  __bf16* W1b   = (__bf16*)d_ws;
  __bf16* W2b   = (__bf16*)((char*)d_ws + 256 * 256 * sizeof(__bf16));
  float*  probs = (float*)((char*)d_ws + 256 * 256 * sizeof(__bf16)
                                        + 256 * 64 * sizeof(__bf16));

  convert_weights_kernel<<<(256 * 256 + 255) / 256, 256, 0, stream>>>(W1, W2, W1b, W2b);

  const int blocks = (ROW_TILES + WAVES_PER_BLOCK - 1) / WAVES_PER_BLOCK;
  mlp_softmax_kernel<<<blocks, 32 * WAVES_PER_BLOCK, 0, stream>>>(
      h0, tab0, tab1, W1b, W2b, b1, b2, probs);

  combine_kernel<<<(NODE_NUM * 64) / 256, 256, 0, stream>>>(probs, idx0, idx1, alpha, out);
}
